// CNN_medium_68444598829697
// MI455X (gfx1250) — 8-bit matrix operand order repaired and checked by reading, compiled, not run on hardware
//
#include <hip/hip_runtime.h>
#include <hip/hip_bf16.h>

typedef int v8i __attribute__((ext_vector_type(8)));
typedef int v4i __attribute__((ext_vector_type(4)));
typedef int v2i __attribute__((ext_vector_type(2)));

#define BN_SLOT(i)   ((size_t)(i) * 4096)
#define WB2_OFF      32768u
#define WB3_OFF      180224u
#define WB4_OFF      475136u
#define WB5_OFF      1064960u
#define WB6_OFF      2244608u
#define WB7_OFF      4603904u
#define ACT_A_OFF    5242880u
#define ACT_B_OFF    43122688u

#define ACT1_BYTES   37879808u
#define ACT2_BYTES   10616832u
#define ACT3_BYTES   21233664u
#define ACT4_BYTES   6553600u
#define ACT5_BYTES   13107200u
#define ACT6_BYTES   2097152u

__device__ __forceinline__ v8i load_a64(const signed char* __restrict__ p, int half) {
    v2i a01 = *(const v2i*)(p +      half * 8);
    v2i a23 = *(const v2i*)(p + 16 + half * 8);
    v2i a45 = *(const v2i*)(p + 32 + half * 8);
    v2i a67 = *(const v2i*)(p + 48 + half * 8);
    v8i a;
    a[0] = a01.x; a[1] = a01.y; a[2] = a23.x; a[3] = a23.y;
    a[4] = a45.x; a[5] = a45.y; a[6] = a67.x; a[7] = a67.y;
    return a;
}

__device__ __forceinline__ v8i load_b64(const signed char* __restrict__ p, int half) {
    v2i b01 = *(const v2i*)(p +      half * 8);
    v2i b23 = *(const v2i*)(p + 16 + half * 8);
    v2i b45 = *(const v2i*)(p + 32 + half * 8);
    v2i b67 = *(const v2i*)(p + 48 + half * 8);
    v8i b;
    b[0] = b01.x; b[1] = b01.y; b[2] = b23.x; b[3] = b23.y;
    b[4] = b45.x; b[5] = b45.y; b[6] = b67.x; b[7] = b67.y;
    return b;
}

__device__ __forceinline__ signed char sign8(float v) {
    return v > 0.f ? (signed char)1 : (v < 0.f ? (signed char)-1 : (signed char)0);
}

__global__ void fill0(int4* __restrict__ p, unsigned n4) {
    unsigned i = blockIdx.x * blockDim.x + threadIdx.x;
    unsigned stride = gridDim.x * blockDim.x;
    int4 z; z.x = 0; z.y = 0; z.z = 0; z.w = 0;
    for (; i < n4; i += stride) p[i] = z;
}

__global__ void prep_bn(const float* __restrict__ g, const float* __restrict__ b,
                        const float* __restrict__ m, const float* __restrict__ v,
                        float* __restrict__ outp, int C) {
    int i = blockIdx.x * blockDim.x + threadIdx.x;
    if (i < C) {
        float sc = g[i] * rsqrtf(v[i] + 1e-5f);
        outp[i]       = sc;
        outp[512 + i] = b[i] - m[i] * sc;
    }
}

__global__ void repack3(const float* __restrict__ w, signed char* __restrict__ wb,
                        int OC, int IC) {
    int idx = blockIdx.x * blockDim.x + threadIdx.x;
    int total = OC * IC * 9;
    if (idx >= total) return;
    int ic = idx % IC;
    int t  = idx / IC;
    int p  = t % 9;
    int oc = t / 9;
    float wv = w[((size_t)(oc * IC + ic)) * 9 + p];
    wb[((size_t)oc * 9 + p) * IC + ic] = sign8(wv);
}

__global__ void repack7(const float* __restrict__ w, signed char* __restrict__ wb) {
    int idx = blockIdx.x * blockDim.x + threadIdx.x;
    if (idx >= 16 * 16 * 512) return;
    int ic = idx & 511;
    int t  = idx >> 9;
    int p  = t & 15;
    int oc = t >> 4;
    signed char s = 0;
    if (oc < 10) s = sign8(w[((size_t)(oc * 512 + ic)) * 16 + p]);
    wb[((size_t)oc * 16 + p) * 512 + ic] = s;
}

__global__ void conv1_bin(const float* __restrict__ x, const float* __restrict__ w1,
                          const float* __restrict__ bnp, signed char* __restrict__ act1) {
    int idx = blockIdx.x * blockDim.x + threadIdx.x;
    if (idx >= 256 * 32 * 32 * 128) return;
    int oc = idx & 127;
    int t  = idx >> 7;
    int xx = t & 31; t >>= 5;
    int yy = t & 31; t >>= 5;
    int n  = t;
    float sum = 0.f;
#pragma unroll
    for (int c = 0; c < 3; ++c)
#pragma unroll
        for (int ky = 0; ky < 3; ++ky)
#pragma unroll
            for (int kx = 0; kx < 3; ++kx) {
                int iy = yy + ky - 1, ix = xx + kx - 1;
                if ((unsigned)iy < 32u && (unsigned)ix < 32u) {
                    float xv = x[(((size_t)n * 3 + c) * 32 + iy) * 32 + ix];
                    float wv = w1[((oc * 3 + c) * 3 + ky) * 3 + kx];
                    float ws = wv > 0.f ? 1.f : (wv < 0.f ? -1.f : 0.f);
                    sum = fmaf(xv, ws, sum);
                }
            }
    float v = sum * bnp[oc] + bnp[512 + oc];
    act1[(((size_t)n * 34 + yy + 1) * 34 + xx + 1) * 128 + oc] = sign8(v);
}

__device__ __forceinline__ void bconv_epilogue(
    const v8i& acc, int oc, int n, int by, int bx, int half,
    int H, int W, int OC, int pool, int outPad,
    const float* __restrict__ bnp, signed char* __restrict__ actOut) {
    float sc = bnp[oc], sh = bnp[512 + oc];
    if (!pool) {
#pragma unroll
        for (int r = 0; r < 8; ++r) {
            int mm = r + 8 * half;
            int yy = by * 4 + (mm >> 2);
            int xx = bx * 4 + (mm & 3);
            float v = (float)acc[r] * sc + sh;
            size_t o = (((size_t)n * (H + 2 * outPad) + yy + outPad) *
                        (W + 2 * outPad) + xx + outPad) * (size_t)OC + oc;
            actOut[o] = sign8(v);
        }
    } else {
        int Ho = H >> 1, Wo = W >> 1;
#pragma unroll
        for (int pxo = 0; pxo < 2; ++pxo) {
            int v0 = max(max(acc[2 * pxo], acc[2 * pxo + 1]),
                         max(acc[2 * pxo + 4], acc[2 * pxo + 5]));
            int yy = by * 2 + half;
            int xx = bx * 2 + pxo;
            float v = (float)v0 * sc + sh;
            size_t o = (((size_t)n * (Ho + 2 * outPad) + yy + outPad) *
                        (Wo + 2 * outPad) + xx + outPad) * (size_t)OC + oc;
            actOut[o] = sign8(v);
        }
    }
}

__global__ void bconv_wmma(const signed char* __restrict__ actIn,
                           const signed char* __restrict__ wb,
                           const float* __restrict__ bnp,
                           signed char* __restrict__ actOut,
                           int N, int H, int W, int C, int OC,
                           int pool, int outPad) {
    int wave = (int)((blockIdx.x * blockDim.x + threadIdx.x) >> 5);
    int lane = threadIdx.x & 31;
    int bw2 = W >> 3;
    int bh  = H >> 2;
    int mtiles = N * bh * bw2;
    int total  = mtiles * (OC >> 5);
    if (wave >= total) return;
    int mt = wave % mtiles;
    int ot = wave / mtiles;
    int bxp = mt % bw2; int t = mt / bw2;
    int by  = t % bh;   int n = t / bh;

    int half = lane >> 4;
    int nl   = lane & 15;
    int py  = by * 4 + (nl >> 2);
    int px0 = bxp * 8 + (nl & 3);
    int Wp  = W + 2;
    int kc  = C >> 6;
    size_t bStride16 = (size_t)16 * 9 * C;

    v8i acc00 = {0, 0, 0, 0, 0, 0, 0, 0};
    v8i acc01 = {0, 0, 0, 0, 0, 0, 0, 0};
    v8i acc10 = {0, 0, 0, 0, 0, 0, 0, 0};
    v8i acc11 = {0, 0, 0, 0, 0, 0, 0, 0};

#pragma unroll
    for (int ky = 0; ky < 3; ++ky)
#pragma unroll
        for (int kx = 0; kx < 3; ++kx) {
            const signed char* ap = actIn +
                (((size_t)n * (H + 2) + (py + ky)) * Wp + (px0 + kx)) * (size_t)C;
            const signed char* bp0 = wb +
                ((size_t)(ot * 32 + nl) * 9 + (ky * 3 + kx)) * (size_t)C;
            const signed char* bp1 = bp0 + bStride16;
            if (ky < 2)
                __builtin_prefetch(ap + (size_t)Wp * C, 0, 0);
            for (int cc = 0; cc < kc; ++cc) {
                v8i a0 = load_a64(ap + cc * 64, half);
                v8i a1 = load_a64(ap + 4 * (size_t)C + cc * 64, half);
                v8i b0 = load_b64(bp0 + cc * 64, half);
                v8i b1 = load_b64(bp1 + cc * 64, half);
                acc00 = __builtin_amdgcn_wmma_i32_16x16x64_iu8(true, a0, true, b0,
                                                               acc00, false, false);
                acc01 = __builtin_amdgcn_wmma_i32_16x16x64_iu8(true, a0, true, b1,
                                                               acc01, false, false);
                acc10 = __builtin_amdgcn_wmma_i32_16x16x64_iu8(true, a1, true, b0,
                                                               acc10, false, false);
                acc11 = __builtin_amdgcn_wmma_i32_16x16x64_iu8(true, a1, true, b1,
                                                               acc11, false, false);
            }
        }

    int oc0 = ot * 32 + nl;
    int oc1 = oc0 + 16;
    int bx0 = bxp * 2;
    int bx1 = bx0 + 1;
    bconv_epilogue(acc00, oc0, n, by, bx0, half, H, W, OC, pool, outPad, bnp, actOut);
    bconv_epilogue(acc01, oc1, n, by, bx0, half, H, W, OC, pool, outPad, bnp, actOut);
    bconv_epilogue(acc10, oc0, n, by, bx1, half, H, W, OC, pool, outPad, bnp, actOut);
    bconv_epilogue(acc11, oc1, n, by, bx1, half, H, W, OC, pool, outPad, bnp, actOut);
}

__global__ void conv7_fc(const signed char* __restrict__ act6,
                         const signed char* __restrict__ wb7,
                         const float* __restrict__ bnp,
                         float* __restrict__ out) {
    int wave = (int)((blockIdx.x * blockDim.x + threadIdx.x) >> 5);
    int lane = threadIdx.x & 31;
    if (wave >= 16) return;
    int half = lane >> 4;
    int nl   = lane & 15;
    int nbase = wave * 16;

    const signed char* ap = act6 + (size_t)(nbase + nl) * 8192;
    const signed char* bp = wb7  + (size_t)nl * 8192;

    v8i acc = {0, 0, 0, 0, 0, 0, 0, 0};
    for (int k = 0; k < 128; ++k) {
        v8i a = load_a64(ap + k * 64, half);
        v8i b = load_b64(bp + k * 64, half);
        acc = __builtin_amdgcn_wmma_i32_16x16x64_iu8(true, a, true, b, acc,
                                                     false, false);
    }

    bool valid = nl < 10;
    float sc = bnp[nl], sh = bnp[512 + nl];
#pragma unroll
    for (int r = 0; r < 8; ++r) {
        int n = nbase + r + 8 * half;
        float v = valid ? fmaf((float)acc[r], sc, sh) : -__builtin_inff();
        float mx = v;
#pragma unroll
        for (int msk = 1; msk < 16; msk <<= 1)
            mx = fmaxf(mx, __shfl_xor(mx, msk, 32));
        float e = expf(v - mx);
        float se = e;
#pragma unroll
        for (int msk = 1; msk < 16; msk <<= 1)
            se += __shfl_xor(se, msk, 32);
        if (valid) out[n * 10 + nl] = v - mx - logf(se);
    }
}

extern "C" void kernel_launch(void* const* d_in, const int* in_sizes, int n_in,
                              void* d_out, int out_size, void* d_ws, size_t ws_size,
                              hipStream_t stream) {
    const float* x  = (const float*)d_in[0];
    const float* w1 = (const float*)d_in[1];
    const float* w2 = (const float*)d_in[2];
    const float* w3 = (const float*)d_in[3];
    const float* w4 = (const float*)d_in[4];
    const float* w5 = (const float*)d_in[5];
    const float* w6 = (const float*)d_in[6];
    const float* w7 = (const float*)d_in[7];
    float* out = (float*)d_out;
    char* ws = (char*)d_ws;

    float* bnp[7];
    for (int i = 0; i < 7; ++i) bnp[i] = (float*)(ws + BN_SLOT(i));
    signed char* wb2 = (signed char*)(ws + WB2_OFF);
    signed char* wb3 = (signed char*)(ws + WB3_OFF);
    signed char* wb4 = (signed char*)(ws + WB4_OFF);
    signed char* wb5 = (signed char*)(ws + WB5_OFF);
    signed char* wb6 = (signed char*)(ws + WB6_OFF);
    signed char* wb7 = (signed char*)(ws + WB7_OFF);
    signed char* actA = (signed char*)(ws + ACT_A_OFF);
    signed char* actB = (signed char*)(ws + ACT_B_OFF);
    signed char* act1 = actA, *act2 = actB, *act3 = actA;
    signed char* act4 = actB, *act5 = actA, *act6 = actB;

    int bnC[7] = {128, 128, 256, 256, 512, 512, 10};
    for (int i = 0; i < 7; ++i)
        prep_bn<<<(bnC[i] + 63) / 64, 64, 0, stream>>>(
            (const float*)d_in[8 + 4 * i], (const float*)d_in[9 + 4 * i],
            (const float*)d_in[10 + 4 * i], (const float*)d_in[11 + 4 * i],
            bnp[i], bnC[i]);

    auto r3 = [&](const float* w, signed char* wb, int OC, int IC) {
        int tot = OC * IC * 9;
        repack3<<<(tot + 255) / 256, 256, 0, stream>>>(w, wb, OC, IC);
    };
    r3(w2, wb2, 128, 128);
    r3(w3, wb3, 256, 128);
    r3(w4, wb4, 256, 256);
    r3(w5, wb5, 512, 256);
    r3(w6, wb6, 512, 512);
    repack7<<<(16 * 16 * 512 + 255) / 256, 256, 0, stream>>>(w7, wb7);

    auto zero = [&](signed char* p, unsigned bytes) {
        unsigned n4 = bytes / 16;
        unsigned blocks = (n4 + 255) / 256;
        if (blocks > 4096) blocks = 4096;
        fill0<<<blocks, 256, 0, stream>>>((int4*)p, n4);
    };
    auto bconv = [&](const signed char* ai, const signed char* wb, const float* bn,
                     signed char* ao, int H, int W, int C, int OC, int pool, int pad) {
        unsigned waves = 256u * (H / 4) * (W / 8) * (OC / 32);
        unsigned blocks = (waves * 32 + 255) / 256;
        bconv_wmma<<<blocks, 256, 0, stream>>>(ai, wb, bn, ao, 256, H, W, C, OC,
                                               pool, pad);
    };

    zero(act1, ACT1_BYTES);
    conv1_bin<<<(256 * 32 * 32 * 128 + 255) / 256, 256, 0, stream>>>(x, w1, bnp[0], act1);

    zero(act2, ACT2_BYTES);
    bconv(act1, wb2, bnp[1], act2, 32, 32, 128, 128,  1,  1);
    zero(act3, ACT3_BYTES);
    bconv(act2, wb3, bnp[2], act3, 16, 16, 128, 256, 0, 1);
    zero(act4, ACT4_BYTES);
    bconv(act3, wb4, bnp[3], act4, 16, 16, 256, 256, 1, 1);
    zero(act5, ACT5_BYTES);
    bconv(act4, wb5, bnp[4], act5, 8, 8, 256, 512, 0, 1);
    bconv(act5, wb6, bnp[5], act6, 8, 8, 512, 512, 1,  0);

    conv7_fc<<<2, 256, 0, stream>>>(act6, wb7, bnp[6], out);
}
